// Yolo_loss_33208687133543
// MI455X (gfx1250) — compile-verified, matrix operand order checked by reading, not run on hardware
//
#include <hip/hip_runtime.h>
#include <hip/hip_fp16.h>

// Problem constants (from reference)
#define RDIM 4096
#define PDIM 32
#define ATTRN 7
#define CDIM (PDIM * ATTRN)   // 224
#define NLAB 32
#define BIMG 64

typedef __attribute__((ext_vector_type(2)))  float    v2f;
typedef __attribute__((ext_vector_type(8)))  float    v8f;
typedef __attribute__((ext_vector_type(16))) _Float16 v16h;

// ---------------------------------------------------------------------------
// Wave32 sum reduction via WMMA ones-matrix trick.
// D = A x Ones + 0 : every column of D is the sum of all A slots.
// Each lane contributes one f32 in A slot (a[0]); a[1] = 0.
// Per-lane: sum its 8 D rows (half a column), then shfl_xor(16) to merge the
// two row halves -> every lane holds the full wave sum, in f32 precision.
// ---------------------------------------------------------------------------
__device__ __forceinline__ float wave_sum32(float v) {
#if __has_builtin(__builtin_amdgcn_wmma_f32_16x16x4_f32)
  v2f a; a[0] = v;    a[1] = 0.0f;
  v2f b; b[0] = 1.0f; b[1] = 1.0f;
  v8f c = {};
  v8f d = __builtin_amdgcn_wmma_f32_16x16x4_f32(
      /*neg_a=*/false, a, /*neg_b=*/false, b,
      /*c_mod=*/(short)0, c, /*reuse_a=*/false, /*reuse_b=*/false);
  float s = ((d[0] + d[1]) + (d[2] + d[3])) + ((d[4] + d[5]) + (d[6] + d[7]));
  s += __shfl_xor(s, 16, 32);
  return s;
#else
  // Fallback: f16 WMMA (codegen-confirmed builtin). A slots distinct per lane,
  // B = ones, f32 accumulate. (f16 rounding of inputs; used only if x4_f32
  // builtin is unavailable.)
  v16h a = {};
  a[0] = (_Float16)v;
  v16h b;
  #pragma unroll
  for (int i = 0; i < 16; ++i) b[i] = (_Float16)1.0f;
  v8f c = {};
  v8f d = __builtin_amdgcn_wmma_f32_16x16x32_f16(
      false, a, false, b, (short)0, c, false, false);
  float s = ((d[0] + d[1]) + (d[2] + d[3])) + ((d[4] + d[5]) + (d[6] + d[7]));
  s += __shfl_xor(s, 16, 32);
  return s;
#endif
}

__device__ __forceinline__ int clampi(int x, int lo, int hi) {
  return x < lo ? lo : (x > hi ? hi : x);
}

__device__ __forceinline__ void label_cell(const float* __restrict__ lab,
                                           int& i, int& j, float& ang) {
  ang = lab[0] + 204.8f;                       // R*CELL_ANGLE/2 = 204.8
  i = clampi((int)floorf(ang / 0.1f), 0, RDIM - 1);
  j = clampi((int)floorf(lab[1]), 0, PDIM - 1); // CELL_DEPTH = 1.0
}

// ---------------------------------------------------------------------------
// Kernel A: l_noobj. Streams ONLY the conf channel (attr 6) for every (b,p)
// row: 33.5 MB total instead of the full 235 MB tensor. One block per (b,p),
// 256 threads x 4 iters x float4 = 4096 r values, fully coalesced.
// Object cells are excluded via a 32-entry membership test held in LDS.
// ---------------------------------------------------------------------------
__global__ __launch_bounds__(256) void yolo_noobj_kernel(
    const float* __restrict__ preds, const float* __restrict__ labels,
    float* __restrict__ partials) {
  const int bp = blockIdx.x;     // 0 .. B*P-1
  const int b  = bp >> 5;
  const int p  = bp & (PDIM - 1);
  const int t  = threadIdx.x;

  __shared__ int   cells[NLAB];
  __shared__ float wsum[8];

  if (t < NLAB) {
    const float* lab = labels + (size_t)(b * NLAB + t) * ATTRN;
    int i, j; float ang;
    label_cell(lab, i, j, ang);
    cells[t] = i * PDIM + j;
  }
  __syncthreads();

  const float* confp =
      preds + (size_t)b * CDIM * RDIM + (size_t)(p * ATTRN + 6) * RDIM;

  float acc = 0.0f;
  #pragma unroll
  for (int it = 0; it < 4; ++it) {
    const int r0 = (it * 256 + t) * 4;
    const float4 x4 = *(const float4*)(confp + r0);
    const float xs[4] = {x4.x, x4.y, x4.z, x4.w};
    #pragma unroll
    for (int u = 0; u < 4; ++u) {
      const int r  = r0 + u;
      const int pc = r * PDIM + p;
      bool obj = false;
      #pragma unroll
      for (int k = 0; k < NLAB; ++k) obj |= (cells[k] == pc);
      // bce(t=0) = -max(log(1 - sigmoid(x)), -100); log(0) -> -inf -> clamp.
      const float s   = 1.0f / (1.0f + __expf(-xs[u]));
      const float bce = -fmaxf(__logf(1.0f - s), -100.0f);
      acc += obj ? 0.0f : bce;
    }
  }

  const float ws = wave_sum32(acc);            // WMMA wave reduction
  if ((t & 31) == 0) wsum[t >> 5] = ws;
  __syncthreads();
  if (t == 0) {
    float s = 0.0f;
    #pragma unroll
    for (int w = 0; w < 8; ++w) s += wsum[w];  // fixed order: deterministic
    partials[bp] = s;
  }
}

// ---------------------------------------------------------------------------
// Kernel B: masked terms (l_xy, l_wl, l_rot, l_obj) at the <=32 object cells
// per image. One wave per image; lane k handles label k. Duplicate (i,j)
// resolved last-wins (lane goes inactive if any later label hits same cell,
// contributing 0 but still participating in the WMMA reductions -> EXEC all 1).
// ---------------------------------------------------------------------------
__global__ __launch_bounds__(32) void yolo_obj_kernel(
    const float* __restrict__ preds, const float* __restrict__ labels,
    float* __restrict__ partialsB) {
  const int b = blockIdx.x;
  const int k = threadIdx.x;

  __shared__ int cells[NLAB];

  const float* lab = labels + (size_t)(b * NLAB + k) * ATTRN;
  const float a1 = lab[1], w = lab[2], l = lab[3], r0 = lab[4], r1 = lab[5];
  int i, j; float ang;
  label_cell(lab, i, j, ang);
  cells[k] = i * PDIM + j;
  __syncthreads();

  bool active = true;
  for (int kk = k + 1; kk < NLAB; ++kk)
    if (cells[kk] == cells[k]) active = false;   // later label overwrites

  float lxy = 0.0f, lwl = 0.0f, lrot = 0.0f, lobj = 0.0f;
  if (active) {
    const float* base =
        preds + (size_t)b * CDIM * RDIM + (size_t)(j * ATTRN) * RDIM + i;
    const float p0 = base[0 * RDIM], p1 = base[1 * RDIM];
    const float p2 = base[2 * RDIM], p3 = base[3 * RDIM];
    const float p4 = base[4 * RDIM], p5 = base[5 * RDIM];
    const float p6 = base[6 * RDIM];

    const float tx = ang / 0.1f - (float)i;
    const float ty = a1 - (float)j;
    const float s0 = 1.0f / (1.0f + __expf(-p0));
    const float s1 = 1.0f / (1.0f + __expf(-p1));
    lxy = (s0 - tx) * (s0 - tx) + (s1 - ty) * (s1 - ty);

    const float tw = __logf(w / 1.6f + 1e-16f);  // ANCHORS = {1.6, 3.9}
    const float tl = __logf(l / 3.9f + 1e-16f);
    lwl = (p2 - tw) * (p2 - tw) + (p3 - tl) * (p3 - tl);

    const float h0 = tanhf(p4), h1 = tanhf(p5);
    lrot = (h0 - r0) * (h0 - r0) + (h1 - r1) * (h1 - r1);

    const float s6 = 1.0f / (1.0f + __expf(-p6));
    lobj = -fmaxf(__logf(s6), -100.0f);          // bce(t=1)
  }

  // All 32 lanes reconverged here -> EXEC all 1 for WMMA.
  const float sxy  = wave_sum32(lxy);
  const float swl  = wave_sum32(lwl);
  const float srot = wave_sum32(lrot);
  const float sobj = wave_sum32(lobj);
  if (k == 0) {
    partialsB[b * 4 + 0] = sxy;
    partialsB[b * 4 + 1] = swl;
    partialsB[b * 4 + 2] = srot;
    partialsB[b * 4 + 3] = sobj;
  }
}

// ---------------------------------------------------------------------------
// Kernel C: deterministic finalize. Fixed-order reductions, writes 6 outputs:
// (total, l_xy, l_wl, l_rot, l_obj, l_noobj).
// ---------------------------------------------------------------------------
__global__ __launch_bounds__(256) void yolo_finalize_kernel(
    const float* __restrict__ pA, const float* __restrict__ pB,
    float* __restrict__ out) {
  __shared__ float sm[256];
  const int t = threadIdx.x;
  float s = 0.0f;
  #pragma unroll
  for (int it = 0; it < 8; ++it) s += pA[it * 256 + t];
  sm[t] = s;
  __syncthreads();
  if (t == 0) {
    float noobj = 0.0f;
    for (int u = 0; u < 256; ++u) noobj += sm[u];
    float lxy = 0.0f, lwl = 0.0f, lrot = 0.0f, lobj = 0.0f;
    for (int b = 0; b < BIMG; ++b) {
      lxy  += pB[b * 4 + 0];
      lwl  += pB[b * 4 + 1];
      lrot += pB[b * 4 + 2];
      lobj += pB[b * 4 + 3];
    }
    const float total = 10.0f * lxy + 10.0f * lwl + 20.0f * lrot +
                        20.0f * lobj + 1.0f * noobj;
    out[0] = total; out[1] = lxy; out[2] = lwl;
    out[3] = lrot;  out[4] = lobj; out[5] = noobj;
  }
}

extern "C" void kernel_launch(void* const* d_in, const int* in_sizes, int n_in,
                              void* d_out, int out_size, void* d_ws, size_t ws_size,
                              hipStream_t stream) {
  const float* preds  = (const float*)d_in[0];   // (64, 224, 4096) f32
  const float* labels = (const float*)d_in[1];   // (64, 32, 7)    f32
  float* ws = (float*)d_ws;
  float* pA = ws;                 // 2048 per-(b,p) noobj partials
  float* pB = ws + BIMG * PDIM;   // 64 x 4 obj-term partials

  yolo_noobj_kernel<<<BIMG * PDIM, 256, 0, stream>>>(preds, labels, pA);
  yolo_obj_kernel<<<BIMG, 32, 0, stream>>>(preds, labels, pB);
  yolo_finalize_kernel<<<1, 256, 0, stream>>>(pA, pB, (float*)d_out);
}
